// Block_5153960755422
// MI455X (gfx1250) — matrix operand order repaired and checked by reading, not compiled, not run on hardware
//
#include <hip/hip_runtime.h>
#include <hip/hip_bf16.h>
#include <math.h>

typedef __attribute__((ext_vector_type(16))) __bf16 v16bf;
typedef __attribute__((ext_vector_type(8)))  float  v8f;

#define LDSPAD 40   // LDS row pitch (elements): 80B rows, 16B aligned, conflict-skewed

union FragU { v16bf f; uint4 q[2]; };

static __device__ __forceinline__ v16bf load_frag2(const __bf16* p0, const __bf16* p1) {
    FragU u;
    u.q[0] = *(const uint4*)p0;
    u.q[1] = *(const uint4*)p1;
    return u.f;
}

static __device__ __forceinline__ v8f wmma_bf16(v16bf a, v16bf b, v8f c) {
    // D = A(16x32 bf16) * B(32x16 bf16) + C(16x16 f32)
    return __builtin_amdgcn_wmma_f32_16x16x32_bf16(false, a, false, b, (short)0, c, false, false);
}

// ---- CDNA5 async global->LDS path (ASYNCcnt), guarded ----
#if __has_builtin(__builtin_amdgcn_global_load_async_to_lds_b128) && \
    __has_builtin(__builtin_amdgcn_s_wait_asynccnt)
#define HAVE_ASYNC_LDS 1
typedef int v4i_t __attribute__((ext_vector_type(4)));
typedef __attribute__((address_space(1))) v4i_t* gptr_v4i;   // global <4 x i32>*
typedef __attribute__((address_space(3))) v4i_t* lptr_v4i;   // LDS    <4 x i32>*
static __device__ __forceinline__ void async_b128(const void* g, void* l) {
    // per-lane 16B copy: LDS[laddr] = MEM[gaddr]; tracked by ASYNCcnt
    __builtin_amdgcn_global_load_async_to_lds_b128(
        (gptr_v4i)(uintptr_t)g,
        (lptr_v4i)(unsigned)(uintptr_t)l,  // flat low 32 bits == LDS offset
        0, 0);
}
#else
#define HAVE_ASYNC_LDS 0
#endif

// ---- cross-lane max over each 16-lane half (VALU permlane path, wave32) ----
#if __has_builtin(__builtin_amdgcn_permlane16)
static __device__ __forceinline__ float plane16(float v, unsigned s0, unsigned s1) {
    int i = __float_as_int(v);
    i = __builtin_amdgcn_permlane16(i, i, s0, s1, false, false);
    return __int_as_float(i);
}
static __device__ __forceinline__ float rowmax16(float v) {
    v = fmaxf(v, plane16(v, 0x67452301u, 0xEFCDAB89u));  // xor 1
    v = fmaxf(v, plane16(v, 0x54761032u, 0xDCFE98BAu));  // xor 2
    v = fmaxf(v, plane16(v, 0x32107654u, 0xBA98FEDCu));  // xor 4
    v = fmaxf(v, plane16(v, 0xFEDCBA98u, 0x76543210u));  // xor 8
    return v;
}
#else
static __device__ __forceinline__ float rowmax16(float v) {
    #pragma unroll
    for (int off = 1; off <= 8; off <<= 1) v = fmaxf(v, __shfl_xor(v, off, 32));
    return v;
}
#endif

// ---------------------------------------------------------------------------
// fp32 [K][N] -> bf16 transposed [N][K]  (one-time weight prep, coalesced writes)
// ---------------------------------------------------------------------------
__global__ void cast_transpose_f32_to_bf16(const float* __restrict__ in,
                                           __bf16* __restrict__ outT, int K, int N) {
    int o = blockIdx.x * blockDim.x + threadIdx.x;   // index into outT [N][K]
    if (o < N * K) {
        int n = o / K, k = o - n * K;
        outT[o] = (__bf16)in[(long)k * N + n];
    }
}

// ---------------------------------------------------------------------------
// Row LayerNorm over C=768, writes bf16. One block (256 thr) per row.
// ---------------------------------------------------------------------------
__global__ __launch_bounds__(256) void layernorm_bf16(
    const float* __restrict__ x, const float* __restrict__ g, const float* __restrict__ bb,
    __bf16* __restrict__ out)
{
    const int row = blockIdx.x;
    const int tid = threadIdx.x;
    const float* xr = x + (long)row * 768;

    float s = 0.f, ss = 0.f;
    #pragma unroll
    for (int c = tid; c < 768; c += 256) { float v = xr[c]; s += v; ss += v * v; }

    __shared__ float r1[256], r2[256];
    r1[tid] = s; r2[tid] = ss;
    __syncthreads();
    for (int off = 128; off > 0; off >>= 1) {
        if (tid < off) { r1[tid] += r1[tid + off]; r2[tid] += r2[tid + off]; }
        __syncthreads();
    }
    const float mu  = r1[0] * (1.0f / 768.0f);
    const float var = r2[0] * (1.0f / 768.0f) - mu * mu;
    const float inv = rsqrtf(var + 1e-6f);

    __bf16* orow = out + (long)row * 768;
    #pragma unroll
    for (int c = tid; c < 768; c += 256)
        orow[c] = (__bf16)((xr[c] - mu) * inv * g[c] + bb[c]);
}

// ---------------------------------------------------------------------------
// WMMA bf16 GEMM: C[M,N] = A[M,K](bf16) @ B[K,N] where B is given TRANSPOSED
// as Bt[N][K]. Block tile 128x128, 8 waves of 64x32, K-step 32, double-buffered
// LDS. A/B tiles staged via async global->LDS copies when available.
// MODE 1: f32 store = acc + bias[col] + res[row,col]
// MODE 2: bf16 store = gelu_exact(acc + bias[col])
// MODE 3: QKV split store: cols [0,1536) -> bf16 C[row,col];
//         cols [1536,2304) (= V) -> vt[b][h][d][n] transposed bf16
// ---------------------------------------------------------------------------
template<int MODE>
__global__ __launch_bounds__(256) void gemm_wmma_bf16(
    const __bf16* __restrict__ A, const __bf16* __restrict__ Bt,
    void* __restrict__ Cout, const float* __restrict__ bias,
    const float* __restrict__ res, __bf16* __restrict__ vt, int K, int N)
{
    __shared__ __bf16 As[2][128 * LDSPAD];   // [row][k], K contiguous
    __shared__ __bf16 Bs[2][128 * LDSPAD];   // [col][k], K contiguous

    const int tid  = threadIdx.x;
    const int lane = tid & 31;
    const int wave = tid >> 5;
    const int wrow = wave >> 2;                 // 0..1  -> 64-row slab
    const int wcol = wave & 3;                  // 0..3  -> 32-col slab
    const int m0   = blockIdx.y * 128;
    const int n0   = blockIdx.x * 128;

    v8f acc[4][2];
    #pragma unroll
    for (int i = 0; i < 4; ++i)
        #pragma unroll
        for (int j = 0; j < 2; ++j)
            #pragma unroll
            for (int e = 0; e < 8; ++e) acc[i][j][e] = 0.f;

    const int rl = lane & 15;          // row/col within 16-tile
    const int hi = lane >> 4;          // lane half
    const int kb  = hi * 8;            // unused (A frags also use kb2)
    const int kb2 = hi * 16;           // B-frag K base

    // staging index precompute (4 x 16B per thread per K-step)
    const int ar  = tid >> 2;          // 0..63  (A row, +64 on second pass)
    const int akc = (tid & 3) * 8;     // 0,8,16,24
    const int bc  = tid >> 1;          // 0..127 (B col)
    const int bkc = (tid & 1) * 16;    // 0,16

    auto stage = [&](int buf, int k0) {
#if HAVE_ASYNC_LDS
        async_b128(A  + (long)(m0 + ar)      * K + k0 + akc,     &As[buf][ar        * LDSPAD + akc]);
        async_b128(A  + (long)(m0 + 64 + ar) * K + k0 + akc,     &As[buf][(64 + ar) * LDSPAD + akc]);
        async_b128(Bt + (long)(n0 + bc)      * K + k0 + bkc,     &Bs[buf][bc        * LDSPAD + bkc]);
        async_b128(Bt + (long)(n0 + bc)      * K + k0 + bkc + 8, &Bs[buf][bc        * LDSPAD + bkc + 8]);
#else
        uint4 a0 = *(const uint4*)(A  + (long)(m0 + ar)      * K + k0 + akc);
        uint4 a1 = *(const uint4*)(A  + (long)(m0 + 64 + ar) * K + k0 + akc);
        uint4 b0 = *(const uint4*)(Bt + (long)(n0 + bc)      * K + k0 + bkc);
        uint4 b1 = *(const uint4*)(Bt + (long)(n0 + bc)      * K + k0 + bkc + 8);
        *(uint4*)(&As[buf][ar        * LDSPAD + akc])     = a0;
        *(uint4*)(&As[buf][(64 + ar) * LDSPAD + akc])     = a1;
        *(uint4*)(&Bs[buf][bc        * LDSPAD + bkc])     = b0;
        *(uint4*)(&Bs[buf][bc        * LDSPAD + bkc + 8]) = b1;
#endif
    };

    const int nk = K >> 5;
    stage(0, 0);
#if HAVE_ASYNC_LDS
    __builtin_amdgcn_s_wait_asynccnt(0);
#endif
    __syncthreads();

    for (int kt = 0; kt < nk; ++kt) {
        const int cur = kt & 1;
#if HAVE_ASYNC_LDS
        // issue next tile's async copies, then wait for *previous* group
        // (async ops retire in order: cnt<=4 => cur buffer resident)
        if (kt + 1 < nk) {
            stage(cur ^ 1, (kt + 1) * 32);
            __builtin_amdgcn_s_wait_asynccnt(4);
        } else {
            __builtin_amdgcn_s_wait_asynccnt(0);
        }
        __syncthreads();   // all threads' cur-buffer copies visible
#else
        if (kt + 1 < nk) stage(cur ^ 1, (kt + 1) * 32);
#endif

        v16bf bfr[2];
        #pragma unroll
        for (int j = 0; j < 2; ++j) {
            const __bf16* pb = &Bs[cur][(wcol * 32 + j * 16 + rl) * LDSPAD + kb2];
            bfr[j] = load_frag2(pb, pb + 8);
        }
        #pragma unroll
        for (int i = 0; i < 4; ++i) {
            const __bf16* pa = &As[cur][(wrow * 64 + i * 16 + rl) * LDSPAD];
            v16bf afr = load_frag2(pa + kb2, pa + kb2 + 8);
            acc[i][0] = wmma_bf16(afr, bfr[0], acc[i][0]);
            acc[i][1] = wmma_bf16(afr, bfr[1], acc[i][1]);
        }
        __syncthreads();   // protect cur buffer before next iteration overwrites
    }

    // ---- epilogue ----
    #pragma unroll
    for (int i = 0; i < 4; ++i)
        #pragma unroll
        for (int j = 0; j < 2; ++j)
            #pragma unroll
            for (int jj = 0; jj < 8; ++jj) {
                const int row = m0 + wrow * 64 + i * 16 + hi * 8 + jj;
                const int col = n0 + wcol * 32 + j * 16 + rl;
                const float v = acc[i][j][jj];
                if (MODE == 1) {
                    ((float*)Cout)[(long)row * N + col] =
                        v + bias[col] + res[(long)row * N + col];
                } else if (MODE == 2) {
                    const float t = v + bias[col];
                    const float gl = 0.5f * t * (1.0f + erff(t * 0.70710678118654752f));
                    ((__bf16*)Cout)[(long)row * N + col] = (__bf16)gl;
                } else { // MODE 3: QKV
                    if (col < 1536) {
                        ((__bf16*)Cout)[(long)row * N + col] = (__bf16)v;
                    } else {
                        const int hc = col - 1536;
                        const int h = hc >> 6, d = hc & 63;
                        const int b = row >> 10, n = row & 1023;
                        vt[(((long)b * 12 + h) * 64 + d) * 1024 + n] = (__bf16)v;
                    }
                }
            }
}

// ---------------------------------------------------------------------------
// Flash attention, one wave per 16-query tile of one (b,h).
// qkv: bf16 [8192, 2304]  (q cols [h*64), k cols [768+h*64); v cols unused)
// vt : bf16 [b][h][64 d][1024 n]  (V pre-transposed by QKV epilogue)
// out: bf16 [8192, 768]
// Per key block: 4 WMMA (QK^T) + 1 WMMA (P row-sums vs all-ones B) + 4 WMMA (PV)
// ---------------------------------------------------------------------------
__global__ __launch_bounds__(256) void attn_flash_wmma(
    const __bf16* __restrict__ qkv, const __bf16* __restrict__ vt,
    __bf16* __restrict__ attn_out)
{
    __shared__ __bf16 Pbuf[8][16 * LDSPAD];   // per-wave: P tile [qrow][key0..31]

    const int tid  = threadIdx.x;
    const int lane = tid & 31;
    const int wave = tid >> 5;
    const int task = blockIdx.x * 8 + wave;   // 6144 tasks total
    const int bh = task >> 6, qt = task & 63;
    const int b = bh / 12, h = bh % 12;
    const int rowQ = b * 1024 + qt * 16;

    const int rl = lane & 15;
    const int hi = lane >> 4;
    const int kb = hi * 8, kb2 = hi * 16;

    // All-ones B fragment: row-sum WMMA operand (bf16 1.0 = 0x3F80)
    FragU uo;
    uo.q[0] = make_uint4(0x3F803F80u, 0x3F803F80u, 0x3F803F80u, 0x3F803F80u);
    uo.q[1] = uo.q[0];
    const v16bf vones = uo.f;

    // Q fragments (d = 0..31, 32..63), resident; fold in softmax scale 1/8
    // (bf16 * 0.125 is an exact exponent shift -> no precision loss)
    const __bf16* qbase = qkv + (long)rowQ * 2304 + h * 64;
    v16bf aq[2];
    #pragma unroll
    for (int s = 0; s < 2; ++s) {
        const __bf16* p = qbase + (long)rl * 2304 + s * 32;
        aq[s] = load_frag2(p + kb2, p + kb2 + 8);
        #pragma unroll
        for (int e = 0; e < 16; ++e) aq[s][e] = aq[s][e] * (__bf16)0.125f;
    }

    float mrow[8], lrow[8], alpha[8];
    v8f o[4];
    #pragma unroll
    for (int jj = 0; jj < 8; ++jj) { mrow[jj] = -1e30f; lrow[jj] = 0.f; }
    #pragma unroll
    for (int t = 0; t < 4; ++t)
        #pragma unroll
        for (int e = 0; e < 8; ++e) o[t][e] = 0.f;

    __bf16* Pw = Pbuf[wave];
    const __bf16* vthead = vt + ((long)b * 12 + h) * 64 * 1024;

    for (int kt = 0; kt < 32; ++kt) {
        const int key0 = kt * 32;

        // ---- S(16x32) = (Q*scale) @ K^T : K frags straight from global ----
        v8f s0, s1;
        #pragma unroll
        for (int e = 0; e < 8; ++e) { s0[e] = 0.f; s1[e] = 0.f; }
        const __bf16* kbase = qkv + (long)(b * 1024 + key0) * 2304 + 768 + h * 64;
        #pragma unroll
        for (int s = 0; s < 2; ++s) {
            const __bf16* p0 = kbase + (long)rl * 2304 + s * 32 + kb2;
            const __bf16* p1 = kbase + (long)(16 + rl) * 2304 + s * 32 + kb2;
            v16bf bk0 = load_frag2(p0, p0 + 8);
            v16bf bk1 = load_frag2(p1, p1 + 8);
            s0 = wmma_bf16(aq[s], bk0, s0);
            s1 = wmma_bf16(aq[s], bk1, s1);
        }

        // ---- online softmax: permlane max-reduce, exp, rescale O, stash P ----
        #pragma unroll
        for (int jj = 0; jj < 8; ++jj) {
            const float x0 = s0[jj];
            const float x1 = s1[jj];
            const float t = rowmax16(fmaxf(x0, x1));
            const float mnew = fmaxf(mrow[jj], t);
            alpha[jj] = __expf(mrow[jj] - mnew);
            mrow[jj] = mnew;
            const float p0 = __expf(x0 - mnew);
            const float p1 = __expf(x1 - mnew);
            #pragma unroll
            for (int t4 = 0; t4 < 4; ++t4) o[t4][jj] *= alpha[jj];
            const int r = hi * 8 + jj;
            Pw[r * LDSPAD + rl]      = (__bf16)p0;
            Pw[r * LDSPAD + 16 + rl] = (__bf16)p1;
        }

        // ---- P fragment (LDS round-trip reformats C-layout -> A-layout) ----
        const __bf16* pp = &Pw[rl * LDSPAD];
        v16bf ap = load_frag2(pp + kb2, pp + kb2 + 8);

        // ---- row sums l_blk = P @ ones via WMMA (replicated across cols) ----
        v8f z;
        #pragma unroll
        for (int e = 0; e < 8; ++e) z[e] = 0.f;
        v8f ls = wmma_bf16(ap, vones, z);
        #pragma unroll
        for (int jj = 0; jj < 8; ++jj) lrow[jj] = lrow[jj] * alpha[jj] + ls[jj];

        // ---- O(16x64) += P(16x32) @ V(32x64): V frags direct from vt ----
        #pragma unroll
        for (int t4 = 0; t4 < 4; ++t4) {
            const __bf16* pv = vthead + (long)(t4 * 16 + rl) * 1024 + key0 + kb2;
            v16bf bv = load_frag2(pv, pv + 8);
            o[t4] = wmma_bf16(ap, bv, o[t4]);
        }
    }

    // ---- normalize + store bf16 ----
    #pragma unroll
    for (int t4 = 0; t4 < 4; ++t4)
        #pragma unroll
        for (int jj = 0; jj < 8; ++jj) {
            const int row = rowQ + hi * 8 + jj;
            const int col = h * 64 + t4 * 16 + rl;
            attn_out[(long)row * 768 + col] = (__bf16)(o[t4][jj] / lrow[jj]);
        }
}

// ---------------------------------------------------------------------------
// Launch: LN1 -> QKV gemm(+V^T) -> flash attn -> proj(+res) -> LN2 -> FC1(gelu)
//         -> FC2(+res)
// ---------------------------------------------------------------------------
extern "C" void kernel_launch(void* const* d_in, const int* in_sizes, int n_in,
                              void* d_out, int out_size, void* d_ws, size_t ws_size,
                              hipStream_t stream)
{
    const float* x      = (const float*)d_in[0];
    const float* qkv_w  = (const float*)d_in[1];
    const float* proj_w = (const float*)d_in[2];
    const float* proj_b = (const float*)d_in[3];
    const float* ln1_g  = (const float*)d_in[4];
    const float* ln1_b  = (const float*)d_in[5];
    const float* ln2_g  = (const float*)d_in[6];
    const float* ln2_b  = (const float*)d_in[7];
    const float* fc1_w  = (const float*)d_in[8];
    const float* fc1_b  = (const float*)d_in[9];
    const float* fc2_w  = (const float*)d_in[10];
    const float* fc2_b  = (const float*)d_in[11];
    float* out = (float*)d_out;

    const int M = 8192;               // B*N tokens
    char* wsp = (char*)d_ws;
    size_t off = 0;
    auto alloc = [&](size_t bytes) -> void* {
        void* p = wsp + off;
        off += (bytes + 255) & ~(size_t)255;
        return p;
    };
    __bf16* qkvwT  = (__bf16*)alloc((size_t)2304 * 768 * 2);  // [N][K]
    __bf16* projwT = (__bf16*)alloc((size_t)768  * 768 * 2);
    __bf16* fc1wT  = (__bf16*)alloc((size_t)3072 * 768 * 2);
    __bf16* fc2wT  = (__bf16*)alloc((size_t)768 * 3072 * 2);
    __bf16* h_bf   = (__bf16*)alloc((size_t)M * 768  * 2);
    __bf16* qkv_bf = (__bf16*)alloc((size_t)M * 2304 * 2);
    __bf16* vt     = (__bf16*)alloc((size_t)M * 768  * 2);    // [b][h][d][n]
    __bf16* attn_bf= (__bf16*)alloc((size_t)M * 768  * 2);
    float*  x2     = (float*) alloc((size_t)M * 768  * 4);
    __bf16* h2_bf  = (__bf16*)alloc((size_t)M * 768  * 2);
    __bf16* h3_bf  = (__bf16*)alloc((size_t)M * 3072 * 2);
    (void)in_sizes; (void)n_in; (void)out_size; (void)ws_size;

    // weight prep: cast + transpose -> Wt[N][K] bf16
    {
        int n;
        n = 2304 * 768; cast_transpose_f32_to_bf16<<<(n + 255) / 256, 256, 0, stream>>>(qkv_w,  qkvwT,  768,  2304);
        n = 768  * 768; cast_transpose_f32_to_bf16<<<(n + 255) / 256, 256, 0, stream>>>(proj_w, projwT, 768,  768);
        n = 3072 * 768; cast_transpose_f32_to_bf16<<<(n + 255) / 256, 256, 0, stream>>>(fc1_w,  fc1wT,  768,  3072);
        n = 768 * 3072; cast_transpose_f32_to_bf16<<<(n + 255) / 256, 256, 0, stream>>>(fc2_w,  fc2wT,  3072, 768);
    }

    // LN1
    layernorm_bf16<<<M, 256, 0, stream>>>(x, ln1_g, ln1_b, h_bf);
    // QKV = h @ qkv_w (M=8192, K=768, N=2304); V written transposed to vt
    gemm_wmma_bf16<3><<<dim3(18, 64), 256, 0, stream>>>(h_bf, qkvwT, qkv_bf, nullptr, nullptr, vt, 768, 2304);
    // Flash attention (6144 wave-tasks, 8 per block)
    attn_flash_wmma<<<768, 256, 0, stream>>>(qkv_bf, vt, attn_bf);
    // x2 = x + attn @ proj_w + proj_b (K=768, N=768)
    gemm_wmma_bf16<1><<<dim3(6, 64), 256, 0, stream>>>(attn_bf, projwT, x2, proj_b, x, nullptr, 768, 768);
    // LN2
    layernorm_bf16<<<M, 256, 0, stream>>>(x2, ln2_g, ln2_b, h2_bf);
    // h3 = gelu(h2 @ fc1_w + fc1_b) (K=768, N=3072)
    gemm_wmma_bf16<2><<<dim3(24, 64), 256, 0, stream>>>(h2_bf, fc1wT, h3_bf, fc1_b, nullptr, nullptr, 768, 3072);
    // out = x2 + h3 @ fc2_w + fc2_b (K=3072, N=768)
    gemm_wmma_bf16<1><<<dim3(6, 64), 256, 0, stream>>>(h3_bf, fc2wT, out, fc2_b, x2, nullptr, 3072, 768);
}
